// Codebook_59236188946702
// MI455X (gfx1250) — compile-verified, matrix operand order checked by reading, not run on hardware
//
#include <hip/hip_runtime.h>
#include <hip/hip_bf16.h>

#define CODE_DIM 512
#define KCODES   2048
#define NROWS    65536          // 64*32*32
#define M_BLOCK  128            // rows per block (8 waves x 16 rows)
#define THREADS  256
#define NT       4              // N-tiles (of 16 codes) per chunk = 64 codes
#define CHUNK_CODES (16 * NT)
#define NCHUNK   (KCODES / CHUNK_CODES)     // 32
#define NBLOCKS  (NROWS / M_BLOCK)          // 512
#define CHUNK_HALVES (CHUNK_CODES * CODE_DIM)   // 32768 halves = 64 KB
#define BBYTES   (2 * CHUNK_HALVES * 2)         // 128 KB: B double buffer,
                                                // overlaid with A staging

typedef __attribute__((ext_vector_type(16))) _Float16 v16h;
typedef __attribute__((ext_vector_type(8)))  _Float16 v8h;
typedef __attribute__((ext_vector_type(8)))  float    v8f;

__device__ __forceinline__ v8f vzero8() {
    v8f z = {0.f, 0.f, 0.f, 0.f, 0.f, 0.f, 0.f, 0.f};
    return z;
}

__device__ __forceinline__ v16h load_frag(const _Float16* p) {
    const v8h* vp = (const v8h*)p;
    v8h lo = vp[0], hi = vp[1];
    return __builtin_shufflevector(lo, hi,
           0, 1, 2, 3, 4, 5, 6, 7, 8, 9, 10, 11, 12, 13, 14, 15);
}

// Issue one per-lane 16-byte async global->LDS copy (ASYNCcnt-tracked).
__device__ __forceinline__ void async_cp16(unsigned int lds_off,
                                           const void* gsrc) {
    asm volatile("global_load_async_to_lds_b128 %0, %1, off"
                 :: "v"(lds_off),
                    "v"((unsigned long long)(uintptr_t)gsrc)
                 : "memory");
}

// ---------------------------------------------------------------------------
// Kernel 1: codebook f32 -> f16 (scaled by 2048), written in the WMMA B
// fragment-linear layout: [tile n][kk][lane][16 halves]; also ||w_k||^2.
// Fragment layout per ISA 7.12.2 (16-bit): lane<16 holds K {kb..kb+7,
// kb+16..kb+23} with kb = kk*32; lane>=16 same with kb += 8.
// ---------------------------------------------------------------------------
__global__ void __launch_bounds__(256) vq_prep(const float* __restrict__ w,
                                               _Float16* __restrict__ wsw,
                                               float* __restrict__ wsum2) {
    __shared__ float red[256];
    const int code = blockIdx.x;
    const int n  = code >> 4;     // tile index
    const int li = code & 15;     // column-lane within tile
    const float* wr = w + (size_t)code * CODE_DIM;
    float s = 0.f;
    for (int d = threadIdx.x; d < CODE_DIM; d += 256) {
        float v = wr[d];
        s = fmaf(v, v, s);
        _Float16 hv = (_Float16)(v * 2048.0f);
        const int kk = d >> 5;
        const int r  = d & 31;
        const int h    = (r >> 3) & 1;          // lane half-group
        const int lane = li + 16 * h;
        const int j    = (r & 7) + (r >= 16 ? 8 : 0);
        wsw[(((size_t)n * 16 + kk) * 32 + lane) * 16 + j] = hv;
    }
    red[threadIdx.x] = s;
    __syncthreads();
    for (int off = 128; off > 0; off >>= 1) {
        if (threadIdx.x < off) red[threadIdx.x] += red[threadIdx.x + off];
        __syncthreads();
    }
    if (threadIdx.x == 0) wsum2[code] = red[0];
}

// ---------------------------------------------------------------------------
// Kernel 2: 128-row tile per block. A tile held in 128 VGPRs per wave
// (16 x v16h fragments); B chunks async-copied into an LDS double buffer
// that overlays the one-shot A staging region (total LDS ~130 KB ->
// 2 blocks/WGP). Register-rotated B pipeline hides LDS latency.
// ---------------------------------------------------------------------------
__global__ void __launch_bounds__(THREADS) vq_main(
        const float* __restrict__ z, const float* __restrict__ weight,
        const _Float16* __restrict__ wsw, const float* __restrict__ wsum2,
        float* __restrict__ out_code, float* __restrict__ out_idx,
        float* __restrict__ partials) {
    extern __shared__ char smem[];
    _Float16* ash    = (_Float16*)smem;            // A staging (one-shot)
    _Float16* bsh    = (_Float16*)smem;            // B double buffer (overlay)
    int*      idxLDS = (int*)(smem + BBYTES);      // 128 ints
    float*    red    = (float*)(smem + BBYTES + M_BLOCK * 4);

    const int tid     = threadIdx.x;
    const int rowbase = blockIdx.x * M_BLOCK;
    const int lane    = tid & 31;
    const int waveId  = tid >> 5;                  // 8 waves x 16 rows

    // ---- stage z tile into LDS, f32->f16, in A-fragment-linear layout
    for (int q = tid; q < M_BLOCK * CODE_DIM / 8; q += THREADS) {
        const int jh = q & 1;
        const int l  = (q >> 1) & 31;
        const int kk = (q >> 6) & 15;
        const int wv = q >> 10;
        const int row = wv * 16 + (l & 15);
        const int d0  = kk * 32 + 8 * (l >> 4) + (jh ? 16 : 0);
        const float4* zp =
            (const float4*)(z + (size_t)(rowbase + row) * CODE_DIM + d0);
        float4 a0 = zp[0], a1 = zp[1];
        v8h hv;
        hv[0] = (_Float16)a0.x; hv[1] = (_Float16)a0.y;
        hv[2] = (_Float16)a0.z; hv[3] = (_Float16)a0.w;
        hv[4] = (_Float16)a1.x; hv[5] = (_Float16)a1.y;
        hv[6] = (_Float16)a1.z; hv[7] = (_Float16)a1.w;
        *(v8h*)(ash + (size_t)q * 8) = hv;
    }
    __syncthreads();

    // ---- pull this wave's whole A tile into registers (16 x v16h = 128 VGPR)
    v16h afrag[16];
#pragma unroll
    for (int kk = 0; kk < 16; ++kk)
        afrag[kk] = load_frag(ash + (((size_t)waveId * 16 + kk) * 32 + lane) * 16);
    asm volatile("s_wait_dscnt 0x0" ::: "memory");  // reads sampled LDS
    __syncthreads();                                // before overlay is reused

    // ---- prologue: async-copy B chunk 0 into buffer 0 (16 instrs/wave)
    const unsigned int bbase = (unsigned int)(uintptr_t)bsh;
    for (int i = tid; i < CHUNK_HALVES / 8; i += THREADS)
        async_cp16(bbase + (unsigned int)i * 16, wsw + (size_t)i * 8);

    float minval[8];
    int   minidx[8];
#pragma unroll
    for (int r = 0; r < 8; ++r) { minval[r] = 3.4e38f; minidx[r] = 0; }

    for (int nc = 0; nc < NCHUNK; ++nc) {
        // issue next chunk into the other buffer, then wait for current
        if (nc + 1 < NCHUNK) {
            const unsigned int nbuf =
                bbase + (unsigned int)(((nc + 1) & 1) * CHUNK_HALVES * 2);
            const _Float16* gsrc = wsw + (size_t)(nc + 1) * CHUNK_HALVES;
            for (int i = tid; i < CHUNK_HALVES / 8; i += THREADS)
                async_cp16(nbuf + (unsigned int)i * 16, gsrc + (size_t)i * 8);
            asm volatile("s_wait_asynccnt 0x10" ::: "memory");
        } else {
            asm volatile("s_wait_asynccnt 0x0" ::: "memory");
        }
        __syncthreads();   // current buffer ready for every wave

        const _Float16* bbuf = bsh + (size_t)(nc & 1) * CHUNK_HALVES;
        const int nbase = nc * CHUNK_CODES;

        v8f acc[NT];
#pragma unroll
        for (int t = 0; t < NT; ++t) acc[t] = vzero8();

        // register-rotated B pipeline: loads for step kk+1 are issued
        // before the WMMAs of step kk (ds returns in order -> partial waits)
        v16h bcur[NT];
#pragma unroll
        for (int t = 0; t < NT; ++t)
            bcur[t] = load_frag(bbuf + (((size_t)t * 16 + 0) * 32 + lane) * 16);

#pragma unroll
        for (int kk = 0; kk < 16; ++kk) {
            v16h bnxt[NT];
            if (kk < 15) {
#pragma unroll
                for (int t = 0; t < NT; ++t)
                    bnxt[t] = load_frag(
                        bbuf + (((size_t)t * 16 + kk + 1) * 32 + lane) * 16);
            }
#pragma unroll
            for (int t = 0; t < NT; ++t)
                acc[t] = __builtin_amdgcn_wmma_f32_16x16x32_f16(
                            false, afrag[kk], false, bcur[t],
                            (short)0, acc[t], false, false);
            if (kk < 15) {
#pragma unroll
                for (int t = 0; t < NT; ++t) bcur[t] = bnxt[t];
            }
        }

        // score_k = ||w_k||^2 - 2*(z.w_k); W scaled by 2048 => -S/1024
        const int col = lane & 15;
#pragma unroll
        for (int t = 0; t < NT; ++t) {
            const int   cidx = nbase + t * 16 + col;
            const float bias = wsum2[cidx];
#pragma unroll
            for (int r = 0; r < 8; ++r) {
                float score = fmaf(acc[t][r], -(1.0f / 1024.0f), bias);
                if (score < minval[r]) { minval[r] = score; minidx[r] = cidx; }
            }
        }
        __syncthreads();   // done reading this buffer before it is refilled
    }

    // ---- cross-lane argmin within each 16-lane half (rows r / r+8 split)
#pragma unroll
    for (int r = 0; r < 8; ++r) {
        float v  = minval[r];
        int   ix = minidx[r];
#pragma unroll
        for (int m = 1; m < 16; m <<= 1) {
            float ov = __shfl_xor(v, m, 32);
            int   oi = __shfl_xor(ix, m, 32);
            if (ov < v || (ov == v && oi < ix)) { v = ov; ix = oi; }
        }
        minval[r] = v; minidx[r] = ix;
    }
    if ((lane & 15) == 0) {
        const int rb = waveId * 16 + (lane >> 4) * 8;
#pragma unroll
        for (int r = 0; r < 8; ++r) {
            idxLDS[rb + r] = minidx[r];
            out_idx[rowbase + rb + r] = (float)minidx[r];
        }
    }
    __syncthreads();

    // ---- gather codebook rows, write code output, exact f32 loss partial
    float partial = 0.f;
    for (int e = tid; e < M_BLOCK * (CODE_DIM / 4); e += THREADS) {
        const int row = e >> 7;          // CODE_DIM/4 == 128
        const int d4  = e & 127;
        const int idx = idxLDS[row];
        float4 wv = ((const float4*)(weight + (size_t)idx * CODE_DIM))[d4];
        float4 zv = ((const float4*)(z + (size_t)(rowbase + row) * CODE_DIM))[d4];
        ((float4*)(out_code + (size_t)(rowbase + row) * CODE_DIM))[d4] = wv;
        float dx = wv.x - zv.x, dy = wv.y - zv.y, dz = wv.z - zv.z, dw = wv.w - zv.w;
        partial = fmaf(dx, dx, partial);
        partial = fmaf(dy, dy, partial);
        partial = fmaf(dz, dz, partial);
        partial = fmaf(dw, dw, partial);
    }
    red[tid] = partial;
    __syncthreads();
    for (int off = 128; off > 0; off >>= 1) {
        if (tid < off) red[tid] += red[tid + off];
        __syncthreads();
    }
    if (tid == 0) partials[blockIdx.x] = red[0];
}

// ---------------------------------------------------------------------------
// Kernel 3: reduce block partials -> loss = 2 * mean((code - z)^2)
// ---------------------------------------------------------------------------
__global__ void __launch_bounds__(256) vq_loss(const float* __restrict__ partials,
                                               float* __restrict__ loss_out) {
    __shared__ float red[256];
    float s = 0.f;
    for (int i = threadIdx.x; i < NBLOCKS; i += 256) s += partials[i];
    red[threadIdx.x] = s;
    __syncthreads();
    for (int off = 128; off > 0; off >>= 1) {
        if (threadIdx.x < off) red[threadIdx.x] += red[threadIdx.x + off];
        __syncthreads();
    }
    if (threadIdx.x == 0)
        loss_out[0] = 2.0f * red[0] / ((float)NROWS * (float)CODE_DIM);
}

// ---------------------------------------------------------------------------
extern "C" void kernel_launch(void* const* d_in, const int* in_sizes, int n_in,
                              void* d_out, int out_size, void* d_ws, size_t ws_size,
                              hipStream_t stream) {
    (void)in_sizes; (void)n_in; (void)out_size; (void)ws_size;
    const float* z = (const float*)d_in[0];
    const float* w = (const float*)d_in[1];

    float* out      = (float*)d_out;
    float* out_code = out;                                   // N*D floats
    float* loss_out = out + (size_t)NROWS * CODE_DIM;        // 1 float
    float* out_idx  = loss_out + 1;                          // N floats

    char*     ws       = (char*)d_ws;
    _Float16* wsw      = (_Float16*)ws;                                      // 2 MB swizzled
    float*    wsum2    = (float*)(ws + (size_t)KCODES * CODE_DIM * 2);       // 8 KB
    float*    partials = (float*)(ws + (size_t)KCODES * CODE_DIM * 2 + KCODES * 4);

    vq_prep<<<KCODES, 256, 0, stream>>>(w, wsw, wsum2);

    const size_t smem = (size_t)BBYTES + M_BLOCK * 4 + THREADS * 4;
    vq_main<<<NBLOCKS, THREADS, smem, stream>>>(z, w, wsw, wsum2,
                                                out_code, out_idx, partials);

    vq_loss<<<1, 256, 0, stream>>>(partials, loss_out);
}
